// LSTM_67851893342326
// MI455X (gfx1250) — compile-verified, matrix operand order checked by reading, not run on hardware
//
#include <hip/hip_runtime.h>
#include <hip/hip_bf16.h>
#include <stdint.h>

typedef __bf16 bf16_t;
typedef __attribute__((ext_vector_type(16))) __bf16 v16bf;
typedef __attribute__((ext_vector_type(8)))  __bf16 v8bf;
typedef __attribute__((ext_vector_type(8)))  float  v8f;
typedef __attribute__((ext_vector_type(4)))  unsigned v4u;
typedef __attribute__((ext_vector_type(8)))  int      v8i_;
typedef __attribute__((ext_vector_type(4)))  int      v4i_;

#define N_B   64
#define T_S   1024
#define D_I   512
#define H_D   512
#define G4    2048              // 4*H
#define M_ALL (N_B * T_S)       // 65536 rows of the xw GEMM

#define REC_BLOCKS 32
#define JPB (H_D / REC_BLOCKS)  // 16 h-columns owned per recurrent block
#define KPAD 520                // padded LDS row stride (bf16 elems) in lstm_rec

// workspace layout (bytes)
#define WS_BAR_OFF  0                       // 256 B barrier region (memset each launch)
#define WS_HBUF_OFF 256                     // 2 * 64*512 bf16 ping-pong h buffer
#define WS_XW_OFF   (256 * 1024)            // xw bf16: 65536*2048*2 = 256 MB

// ---------------------------------------------------------------------------
// Build one 16x32 bf16 WMMA operand fragment from LDS (ISA 7.12.2 layout).
__device__ inline v16bf lds_frag(const bf16_t* base, int rc0, int kbase,
                                 int kstride, int lane) {
  const int rc = rc0 + (lane & 15);
  const int kh = (lane >> 4) & 1;
  const bf16_t* p = base + rc * kstride + kbase + kh * 8;
  v16bf f;
#pragma unroll
  for (int e = 0; e < 8; ++e) f[e] = p[e];
#pragma unroll
  for (int e = 0; e < 8; ++e) f[8 + e] = p[16 + e];
  return f;
}

// ---------------------------------------------------------------------------
// TDM: stage 64x512 bf16 (contiguous rows) from global into LDS with HW row
// padding: +16B after every 1024B row => LDS row stride 1040B (= KPAD elems).
#if __has_builtin(__builtin_amdgcn_tensor_load_to_lds)
#define HAVE_TDM 1
__device__ inline void tdm_load_h(const bf16_t* hsrc, uint32_t lds_off) {
  const uint64_t ga = (uint64_t)(uintptr_t)hsrc;
  v4u g0;
  g0[0] = 1u;                                           // count=1, user mode
  g0[1] = lds_off;                                      // lds_addr (bytes)
  g0[2] = (uint32_t)(ga & 0xffffffffu);                 // global_addr[31:0]
  g0[3] = (uint32_t)((ga >> 32) & 0x01ffffffu)          // global_addr[56:32]
        | (2u << 30);                                   // type=2 ("image")
  v8i_ g1;
  g1[0] = (int)((1u << 16)      // data_size = 2 bytes
              | (1u << 20)      // pad_enable
              | (7u << 22)      // pad_interval: every 256 DWORDs (1024 B)
              | (3u << 25));    // pad_amount: 4 DWORDs (16 B)
  g1[1] = (int)(512u << 16);    // tensor_dim0[15:0] = 512 @ bits[63:48]
  g1[2] = (int)(64u  << 16);    // tensor_dim0 hi = 0 | tensor_dim1 = 64
  g1[3] = (int)(512u << 16);    // tensor_dim1 hi = 0 | tile_dim0 = 512
  g1[4] = 64;                   // tile_dim1 = 64, tile_dim2 = 0
  g1[5] = 512;                  // tensor_dim0_stride = 512 elems
  g1[6] = 0;
  g1[7] = 0;
  const v4i_ z4 = {0, 0, 0, 0};
#if __has_include(<hip/amd_detail/amd_gfx1250_TDM.h>)
  const v8i_ z8 = {0, 0, 0, 0, 0, 0, 0, 0};
  __builtin_amdgcn_tensor_load_to_lds(g0, g1, z4, z4, z8, 0);   // 6-arg toolchain
#else
  __builtin_amdgcn_tensor_load_to_lds(g0, g1, z4, z4, 0);       // 5-arg ROCm 7.2
#endif
}
#else
#define HAVE_TDM 0
#endif

// ---------------------------------------------------------------------------
// Kernel 1: xw = bf16( x @ Wx + b )   (M_ALL x 2048, K = 512)
// Block tile 128x128, K-step 32, 8 waves; wave tile 32x64 (2x4 WMMA tiles).
__global__ __launch_bounds__(256) void xw_gemm(const float* __restrict__ x,
                                               const float* __restrict__ Wx,
                                               const float* __restrict__ bias,
                                               bf16_t* __restrict__ xw) {
  __shared__ __align__(16) bf16_t At[128 * 40];   // 128 rows x 32 K (pad 40)
  __shared__ __align__(16) bf16_t Bt[128 * 40];   // 128 cols x 32 K (col-major)

  const int tid  = threadIdx.x;
  const int lane = tid & 31, wid = tid >> 5;
  const int wm = wid & 3, wn = wid >> 2;          // 4 waves in M, 2 in N
  const int n0 = blockIdx.x * 128;
  const int m0 = blockIdx.y * 128;

  v8f acc[2][4];
#pragma unroll
  for (int i = 0; i < 2; ++i)
#pragma unroll
    for (int j = 0; j < 4; ++j) acc[i][j] = {};

  for (int k0 = 0; k0 < D_I; k0 += 32) {
    __syncthreads();
    // stage A (x) 128x32: float4 x2 loads -> 8xbf16 packed -> one b128 LDS store
#pragma unroll
    for (int i = tid; i < 128 * 4; i += 256) {
      int r = i >> 2, c8 = (i & 3) << 3;
      const float* gp = &x[(size_t)(m0 + r) * D_I + (k0 + c8)];
      float4 f0 = *(const float4*)gp;
      float4 f1 = *(const float4*)(gp + 4);
      v8bf v;
      v[0] = (bf16_t)f0.x; v[1] = (bf16_t)f0.y;
      v[2] = (bf16_t)f0.z; v[3] = (bf16_t)f0.w;
      v[4] = (bf16_t)f1.x; v[5] = (bf16_t)f1.y;
      v[6] = (bf16_t)f1.z; v[7] = (bf16_t)f1.w;
      *(v8bf*)&At[r * 40 + c8] = v;               // byte off r*80+c8*2, 16B-aligned
    }
    // stage B (Wx) 32x128: float4 loads, transposed bf16 scatter into LDS
#pragma unroll
    for (int i = tid; i < 32 * 32; i += 256) {
      int r = i >> 5, c4 = (i & 31) << 2;
      float4 f = *(const float4*)&Wx[(size_t)(k0 + r) * G4 + (n0 + c4)];
      Bt[(c4 + 0) * 40 + r] = (bf16_t)f.x;
      Bt[(c4 + 1) * 40 + r] = (bf16_t)f.y;
      Bt[(c4 + 2) * 40 + r] = (bf16_t)f.z;
      Bt[(c4 + 3) * 40 + r] = (bf16_t)f.w;
    }
    __syncthreads();

    v16bf af[2], bq[4];
#pragma unroll
    for (int tm = 0; tm < 2; ++tm)
      af[tm] = lds_frag(At, wm * 32 + tm * 16, 0, 40, lane);
#pragma unroll
    for (int tn = 0; tn < 4; ++tn)
      bq[tn] = lds_frag(Bt, wn * 64 + tn * 16, 0, 40, lane);
#pragma unroll
    for (int tm = 0; tm < 2; ++tm)
#pragma unroll
      for (int tn = 0; tn < 4; ++tn)
        acc[tm][tn] = __builtin_amdgcn_wmma_f32_16x16x32_bf16(
            false, af[tm], false, bq[tn], (short)0, acc[tm][tn], false, false);
  }

  // epilogue: C layout — VGPR r holds M = r + 8*(lane>=16), lane&15 = N col
  const int rbase = (lane >> 4) * 8;
  const int cl    = lane & 15;
#pragma unroll
  for (int tm = 0; tm < 2; ++tm)
#pragma unroll
    for (int tn = 0; tn < 4; ++tn) {
      int gm = m0 + wm * 32 + tm * 16 + rbase;
      int gn = n0 + wn * 64 + tn * 16 + cl;
      float bv = bias[gn];
#pragma unroll
      for (int r = 0; r < 8; ++r)
        xw[(size_t)(gm + r) * G4 + gn] = (bf16_t)(acc[tm][tn][r] + bv);
    }
}

// ---------------------------------------------------------------------------
// Kernel 2: init ping-pong h buffer (slot 0) from h0
__global__ void init_h(const float* __restrict__ h0, bf16_t* __restrict__ hbuf) {
  int i = blockIdx.x * blockDim.x + threadIdx.x;
  if (i < N_B * H_D) hbuf[i] = (bf16_t)h0[i];
}

// ---------------------------------------------------------------------------
// Kernel 3: persistent recurrent LSTM.
__global__ __launch_bounds__(256) void lstm_rec(const float* __restrict__ Wh,
                                                const bf16_t* __restrict__ xw,
                                                bf16_t* __restrict__ hbuf,
                                                float* __restrict__ out,
                                                unsigned* __restrict__ bar) {
  extern __shared__ char smem[];
  bf16_t* WhT   = (bf16_t*)smem;                  // 64 cols x KPAD bf16
  bf16_t* hA    = WhT + 64 * KPAD;                // 64 rows x KPAD bf16
  float*  gates = (float*)(hA + 64 * KPAD);       // [4][64][16]
  float*  cst   = gates + 4 * N_B * JPB;          // [64][16]

  const int tid  = threadIdx.x;
  const int lane = tid & 31, wid = tid >> 5;
  const int jblk = blockIdx.x * JPB;

  // one-time: Wh gate strips -> bf16 col-major LDS (resident all T steps)
  for (int i = tid; i < 64 * 512; i += 256) {
    int c = i >> 9, k = i & 511;                  // c = g*16 + jj
    int g = c >> 4, jj = c & 15;
    WhT[c * KPAD + k] = (bf16_t)Wh[(size_t)k * G4 + g * H_D + jblk + jj];
  }
  for (int i = tid; i < N_B * JPB; i += 256) cst[i] = 0.f;
  __syncthreads();

  for (int t = 0; t < T_S; ++t) {
    const bf16_t* hsrc = hbuf + (size_t)(t & 1) * N_B * H_D;
    bf16_t*       hdst = hbuf + (size_t)((t + 1) & 1) * N_B * H_D;

    // stage full h (64x512 bf16) into padded LDS
#if HAVE_TDM
    if (tid == 0) {
      tdm_load_h(hsrc, (uint32_t)(uintptr_t)&hA[0]);
      __builtin_amdgcn_s_wait_tensorcnt((short)0);
    }
#else
    for (int i = tid; i < N_B * (H_D / 2); i += 256) {
      int r = i >> 8, c2 = i & 255;
      *(uint32_t*)&hA[r * KPAD + c2 * 2] = ((const uint32_t*)hsrc)[r * 256 + c2];
    }
#endif
    __syncthreads();

    // h @ Wh (strips): 16 tiles of 16x16, 2 per wave, K = 512 in steps of 32
#pragma unroll
    for (int it = 0; it < 2; ++it) {
      const int id = wid * 2 + it;
      const int tm = id & 3;        // batch tile (4 x 16 rows)
      const int tc = id >> 2;       // gate index 0..3 (i,f,o,g)
      v8f acc = {};
#pragma unroll
      for (int k0 = 0; k0 < H_D; k0 += 32) {
        v16bf a = lds_frag(hA,  tm * 16, k0, KPAD, lane);
        v16bf w = lds_frag(WhT, tc * 16, k0, KPAD, lane);
        acc = __builtin_amdgcn_wmma_f32_16x16x32_bf16(
            false, a, false, w, (short)0, acc, false, false);
      }
      const int rbase = tm * 16 + (lane >> 4) * 8;
      const int col   = lane & 15;
      const int gcol  = tc * H_D + jblk + col;
      if (t + 1 < T_S)   // prefetch next timestep's xw strip (global_prefetch_b8)
        __builtin_prefetch(&xw[((size_t)rbase * T_S + t + 1) * G4 + gcol], 0, 1);
#pragma unroll
      for (int r = 0; r < 8; ++r) {
        int n = rbase + r;
        size_t m = (size_t)n * T_S + t;
        float av = acc[r] + (float)xw[m * G4 + gcol];
        float v = (tc == 3) ? tanhf(av) : 1.f / (1.f + __expf(-av));
        gates[(tc * N_B + n) * JPB + col] = v;
      }
    }
    __syncthreads();

    // elementwise cell/hidden update for owned columns
    for (int i = tid; i < N_B * JPB; i += 256) {
      int n = i >> 4, j = i & 15;
      float iv = gates[(0 * N_B + n) * JPB + j];
      float fv = gates[(1 * N_B + n) * JPB + j];
      float ov = gates[(2 * N_B + n) * JPB + j];
      float gv = gates[(3 * N_B + n) * JPB + j];
      float c  = fv * cst[i] + iv * gv;
      float h  = ov * tanhf(c);
      cst[i] = c;
      out[(size_t)n * T_S * H_D + (size_t)t * H_D + jblk + j] = h;
      hdst[n * H_D + jblk + j] = (bf16_t)h;
    }

    // grid-wide sense-reversing barrier (agent scope, acq/rel)
    __syncthreads();
    if (tid == 0) {
      unsigned g = __hip_atomic_load(&bar[1], __ATOMIC_ACQUIRE,
                                     __HIP_MEMORY_SCOPE_AGENT);
      unsigned arrived = __hip_atomic_fetch_add(&bar[0], 1u, __ATOMIC_ACQ_REL,
                                                __HIP_MEMORY_SCOPE_AGENT) + 1u;
      if (arrived == REC_BLOCKS) {
        __hip_atomic_store(&bar[0], 0u, __ATOMIC_RELAXED,
                           __HIP_MEMORY_SCOPE_AGENT);
        __hip_atomic_fetch_add(&bar[1], 1u, __ATOMIC_ACQ_REL,
                               __HIP_MEMORY_SCOPE_AGENT);
      } else {
        while (__hip_atomic_load(&bar[1], __ATOMIC_ACQUIRE,
                                 __HIP_MEMORY_SCOPE_AGENT) == g) {
          __builtin_amdgcn_s_sleep(2);
        }
      }
    }
    __syncthreads();
  }
}

// ---------------------------------------------------------------------------
extern "C" void kernel_launch(void* const* d_in, const int* in_sizes, int n_in,
                              void* d_out, int out_size, void* d_ws, size_t ws_size,
                              hipStream_t stream) {
  const float* x    = (const float*)d_in[0];   // [64,1024,512]
  const float* h0   = (const float*)d_in[1];   // [64,512]
  const float* Wx   = (const float*)d_in[2];   // [512,2048]
  const float* Wh   = (const float*)d_in[3];   // [512,2048]
  const float* bias = (const float*)d_in[4];   // [2048]
  float* out = (float*)d_out;                  // [64,1024,512]

  unsigned* bar  = (unsigned*)((char*)d_ws + WS_BAR_OFF);
  bf16_t*   hbuf = (bf16_t*)((char*)d_ws + WS_HBUF_OFF);
  bf16_t*   xwb  = (bf16_t*)((char*)d_ws + WS_XW_OFF);

  // reset barrier state (graph-capture-safe)
  hipMemsetAsync((char*)d_ws + WS_BAR_OFF, 0, 256, stream);

  init_h<<<(N_B * H_D + 255) / 256, 256, 0, stream>>>(h0, hbuf);

  dim3 ggrid(G4 / 128, M_ALL / 128);           // (16, 512)
  xw_gemm<<<ggrid, 256, 0, stream>>>(x, Wx, bias, xwb);

  const size_t smem_bytes =
      (size_t)(2 * 64 * KPAD) * sizeof(bf16_t) +             // WhT + hA
      (size_t)(4 * N_B * JPB + N_B * JPB) * sizeof(float);   // gates + c
  lstm_rec<<<REC_BLOCKS, 256, smem_bytes, stream>>>(Wh, xwb, hbuf, out, bar);
}
